// CapsModel_1709396984017
// MI455X (gfx1250) — matrix operand order repaired and checked by reading, not compiled, not run on hardware
//
#include <hip/hip_runtime.h>

typedef __attribute__((ext_vector_type(16))) _Float16 v16h;
typedef __attribute__((ext_vector_type(8)))  _Float16 v8h;
typedef __attribute__((ext_vector_type(8)))  float    v8f;

// ---------------- problem constants ----------------
#define B_    64
#define N_    32
#define Himg  16
#define Kk    3
#define M_    32
#define HO    7
#define NPOS  (B_*HO*HO)      // 3136 spatial positions
#define NKL   (N_*Kk*Kk)      // 288
#define D16   16
#define CLS   10
#define NFC   (M_*HO*HO)      // 1568

// GEMM1: [GM1 x GK1] @ [GK1 x GN1]  (initial conv-caps projection)
#define GM1 (NPOS*4)          // 12544  rows = (b,hw,a)
#define GK1 (NKL*4)           // 1152   k    = (n,k,l,x)
#define GN1 (M_*4)            // 128    cols = (m,d)
// GEMM2: FC caps projection
#define GM2 (B_*4)            // 256    rows = (b,a)
#define GK2 (NFC*4)           // 6272   k    = (n,x)
#define GN2 (CLS*4)           // 40     cols = (m2,d)

// ================= WMMA GEMM (f16 in, f32 out) =================
// C[M,N] = A[M,K] @ B[K,N], row-major. Requires M%128==0, K%32==0 (true
// for both uses); N guarded. Tile 128x128, K-step 32, 8 waves, each wave
// owns a 32x64 patch = 2x4 grid of 16x16 WMMA accumulators.
#define TM 128
#define TN 128
#define TK 32

__global__ __launch_bounds__(256) void wmma_gemm_f16(
    const _Float16* __restrict__ A, const _Float16* __restrict__ B,
    float* __restrict__ C, int M, int N, int K)
{
  __shared__ _Float16 sA [TM][TK + 8];   // row-major A tile
  __shared__ _Float16 sBt[TN][TK + 8];   // transposed B tile

  const int tid  = threadIdx.x;
  const int lane = tid & 31;
  const int wid  = tid >> 5;
  const int wRow = (wid >> 1) * 32;
  const int wCol = (wid & 1) * 64;
  const int rowBase = blockIdx.x * TM;
  const int colBase = blockIdx.y * TN;
  const int l16 = lane & 15;
  const int lhi = lane >> 4;

  v8f acc[2][4];
  v8f zz = {};
  for (int i = 0; i < 2; ++i)
    for (int j = 0; j < 4; ++j) acc[i][j] = zz;

  for (int k0 = 0; k0 < K; k0 += TK) {
    // ---- stage A tile: 128x32 halves in two 16B chunks per thread;
    //      issue both global loads before the LDS stores so the compiler
    //      can stagger s_wait_loadcnt instead of serializing.
    {
      const int c0 = tid, c1 = tid + 256;
      const int r0 = c0 >> 2, kc0 = (c0 & 3) * 8;
      const int r1 = c1 >> 2, kc1 = (c1 & 3) * 8;
      v8h a0 = *(const v8h*)(A + (size_t)(rowBase + r0) * K + k0 + kc0);
      v8h a1 = *(const v8h*)(A + (size_t)(rowBase + r1) * K + k0 + kc1);
      *(v8h*)&sA[r0][kc0] = a0;
      *(v8h*)&sA[r1][kc1] = a1;
    }
    // ---- stage B tile transposed; vector fast path when fully in-bounds
    //      (always for GEMM1; n0 in {0,16} for GEMM2's N=40)
    {
      const int kk = tid >> 3;              // 0..31
      const int n0 = (tid & 7) * 16;        // 0..112
      const _Float16* src = B + (size_t)(k0 + kk) * N + colBase + n0;
      if (colBase + n0 + 16 <= N) {
        v8h b0 = *(const v8h*)(src);
        v8h b1 = *(const v8h*)(src + 8);
#pragma unroll
        for (int j = 0; j < 8; ++j) {
          sBt[n0 + j][kk]     = b0[j];
          sBt[n0 + 8 + j][kk] = b1[j];
        }
      } else {
#pragma unroll
        for (int j = 0; j < 16; ++j)
          sBt[n0 + j][kk] = (colBase + n0 + j < N) ? src[j] : (_Float16)0.f;
      }
    }
    __syncthreads();

    if (k0 + TK < K) {  // hint next K-tile into cache (global_prefetch_b8)
      __builtin_prefetch(A + (size_t)(rowBase + (tid & 127)) * K + k0 + TK, 0, 1);
      __builtin_prefetch(B + (size_t)(k0 + TK + (tid & 31)) * N + colBase, 0, 1);
    }

    // ---- build fragments (A 16x32 f16: lane<16 rows K0-7/16-23, lane>=16 K8-15/24-31;
    //      B 32x16 from transposed LDS: lane n-col, same K split as A)
    v16h af[2], bf[4];
#pragma unroll
    for (int rt = 0; rt < 2; ++rt) {
      int r = wRow + rt*16 + l16;
      v8h lo = *(const v8h*)&sA[r][lhi*8];
      v8h hi = *(const v8h*)&sA[r][lhi*8 + 16];
      af[rt] = __builtin_shufflevector(lo, hi, 0,1,2,3,4,5,6,7,8,9,10,11,12,13,14,15);
    }
#pragma unroll
    for (int ct = 0; ct < 4; ++ct) {
      int n = wCol + ct*16 + l16;
      v8h lo = *(const v8h*)&sBt[n][lhi*8];
      v8h hi = *(const v8h*)&sBt[n][lhi*8 + 16];
      bf[ct] = __builtin_shufflevector(lo, hi, 0,1,2,3,4,5,6,7,8,9,10,11,12,13,14,15);
    }
#pragma unroll
    for (int rt = 0; rt < 2; ++rt)
#pragma unroll
      for (int ct = 0; ct < 4; ++ct)
        acc[rt][ct] = __builtin_amdgcn_wmma_f32_16x16x32_f16(
            false, af[rt], false, bf[ct], (short)0, acc[rt][ct], false, false);
    __syncthreads();
  }

  // ---- store: C/D layout: lane%16 = col, VGPR j = row j + 8*(lane/16)
#pragma unroll
  for (int rt = 0; rt < 2; ++rt)
#pragma unroll
    for (int ct = 0; ct < 4; ++ct) {
      int col  = colBase + wCol + ct*16 + l16;
      int row0 = rowBase + wRow + rt*16 + lhi*8;
      if (col < N) {
#pragma unroll
        for (int j = 0; j < 8; ++j)
          C[(size_t)(row0 + j) * N + col] = acc[rt][ct][j];
      }
    }
}

// ================= layernorm helper =================
__device__ inline void ln16(const float* o, const float* g, const float* bb,
                            float* out)
{
  float mu = 0.f;
#pragma unroll
  for (int i = 0; i < 16; ++i) mu += o[i];
  mu *= (1.f / 16.f);
  float var = 0.f;
#pragma unroll
  for (int i = 0; i < 16; ++i) { float d = o[i] - mu; var += d * d; }
  var *= (1.f / 16.f);
  float r = rsqrtf(var + 1e-5f);
#pragma unroll
  for (int i = 0; i < 16; ++i) out[i] = (o[i] - mu) * r * g[i] + bb[i];
}

// ================= pack kernels (f32 -> f16 GEMM operands) =================
__global__ void pack_A1(const float* __restrict__ X, _Float16* __restrict__ A)
{
  size_t i = (size_t)blockIdx.x * blockDim.x + threadIdx.x;
  if (i >= (size_t)GM1 * GK1) return;
  int r  = (int)(i / GK1);
  int kc = (int)(i % GK1);
  int b = r / (HO*HO*4); int rem = r % (HO*HO*4);
  int hw = rem >> 2;  int a = rem & 3;
  int h = hw / HO, w = hw % HO;
  int n = kc / 36; int r2 = kc % 36;
  int k = r2 / 12; int l = (r2 % 12) / 4; int x = r2 & 3;
  A[i] = (_Float16)X[((((size_t)b*N_ + n)*Himg + (h*2+k))*Himg + (w*2+l))*16 + a*4 + x];
}

__global__ void pack_B1(const float* __restrict__ Wc, _Float16* __restrict__ Bm)
{
  size_t i = (size_t)blockIdx.x * blockDim.x + threadIdx.x;
  if (i >= (size_t)GK1 * GN1) return;
  int kc = (int)(i / GN1);
  int cc = (int)(i % GN1);
  int m = cc >> 2, d = cc & 3;
  int n = kc / 36; int k = (kc % 36) / 12; int l = (kc % 12) / 4; int x = kc & 3;
  Bm[i] = (_Float16)Wc[(((((size_t)k*Kk + l)*N_ + n)*4 + x)*4 + d)*M_ + m];
}

__global__ void pack_A2(const float* __restrict__ vb, _Float16* __restrict__ A)
{
  size_t i = (size_t)blockIdx.x * blockDim.x + threadIdx.x;
  if (i >= (size_t)GM2 * GK2) return;
  int r = (int)(i / GK2); int c = (int)(i % GK2);
  int b = r >> 2, a = r & 3;
  int n = c >> 2, x = c & 3;
  int m = n / (HO*HO); int hw = n % (HO*HO);
  A[i] = (_Float16)vb[((size_t)(b*HO*HO + hw)*M_ + m)*16 + a*4 + x];
}

__global__ void pack_B2(const float* __restrict__ Wfc, _Float16* __restrict__ Bm)
{
  size_t i = (size_t)blockIdx.x * blockDim.x + threadIdx.x;
  if (i >= (size_t)GK2 * GN2) return;
  int kc = (int)(i / GN2); int cc = (int)(i % GN2);
  int n = kc >> 2, x = kc & 3;
  int m2 = cc / 4, d = cc % 4;
  Bm[i] = (_Float16)Wfc[(size_t)n*160 + (x*4 + d)*CLS + m2];
}

// ================= v init: /M + LN =================
__global__ void v_init_ln(const float* __restrict__ O1,
                          const float* __restrict__ g, const float* __restrict__ bb,
                          float* __restrict__ vb)
{
  int idx = blockIdx.x * blockDim.x + threadIdx.x;
  if (idx >= B_ * M_ * HO * HO) return;
  int b = idx / (M_*HO*HO); int m = (idx / (HO*HO)) % M_; int hw = idx % (HO*HO);
  size_t r0 = ((size_t)b*HO*HO + hw) * 4;
  float o[16];
#pragma unroll
  for (int a = 0; a < 4; ++a)
#pragma unroll
    for (int d = 0; d < 4; ++d)
      o[a*4+d] = O1[(r0 + a)*GN1 + m*4 + d] * (1.f / (float)M_);
  float out[16];
  ln16(o, g, bb, out);
  float* vp = vb + ((size_t)(b*HO*HO + hw)*M_ + m)*16;
#pragma unroll
  for (int i = 0; i < 16; ++i) vp[i] = out[i];
}

// ================= u init: /Cls + LN =================
__global__ void u_init_ln(const float* __restrict__ O2,
                          const float* __restrict__ g, const float* __restrict__ bb,
                          float* __restrict__ ub)
{
  int idx = blockIdx.x * blockDim.x + threadIdx.x;
  if (idx >= B_ * CLS) return;
  int b = idx / CLS; int m2 = idx % CLS;
  float o[16];
#pragma unroll
  for (int a = 0; a < 4; ++a)
#pragma unroll
    for (int d = 0; d < 4; ++d)
      o[a*4+d] = O2[(size_t)(b*4 + a)*GN2 + m2*4 + d] * (1.f / (float)CLS);
  float out[16];
  ln16(o, g, bb, out);
#pragma unroll
  for (int i = 0; i < 16; ++i) ub[(size_t)idx*16 + i] = out[i];
}

// ================= fused conv routing round (in-place on vbuf) =============
// One workgroup per (b,h,w). Softmax is over m, local per nkl row, so
// logits never leave LDS. Update reduced with ds_add_f32.
__global__ __launch_bounds__(256) void conv_route(
    const float* __restrict__ X, const float* __restrict__ Wc,
    float* __restrict__ vb,
    const float* __restrict__ g, const float* __restrict__ bb)
{
  __shared__ float sP [NKL*16];   // patches [nkl][a*4+x]
  __shared__ float sQK[NKL*M_];   // coupling [nkl][m]
  __shared__ float sV [M_*16];    // current v [m][a*4+d]
  __shared__ float sVn[M_*16];    // updated v (atomic accum)

  const int tid = threadIdx.x;
  const int pos = blockIdx.x;              // b*49+hw
  const int b = pos / (HO*HO); const int hw = pos % (HO*HO);
  const int h = hw / HO, w = hw % HO;

  for (int i = tid; i < NKL*16; i += 256) {
    int nkl = i >> 4; int t = i & 15;
    int n = nkl / 9; int kl = nkl % 9; int k = kl / 3; int l = kl % 3;
    sP[i] = X[((((size_t)b*N_ + n)*Himg + (h*2+k))*Himg + (w*2+l))*16 + t];
  }
  sV[tid]       = vb[(size_t)pos*512 + tid];
  sV[tid + 256] = vb[(size_t)pos*512 + tid + 256];
  __syncthreads();

  // ---- phase 1: logits -> softmax(m) -> normalized coupling
  for (int nkl = tid; nkl < NKL; nkl += 256) {
    float p[16];
#pragma unroll
    for (int i = 0; i < 16; ++i) p[i] = sP[nkl*16 + i];
    int n = nkl / 9; int kl = nkl % 9;
    const float* wp = Wc + ((size_t)kl*N_ + n)*512;   // + (x*4+d)*32 + m
    float lg[32];
    for (int m = 0; m < M_; ++m) {
      float acc = 0.f;
#pragma unroll
      for (int x = 0; x < 4; ++x)
#pragma unroll
        for (int d = 0; d < 4; ++d) {
          float s = 0.f;
#pragma unroll
          for (int a = 0; a < 4; ++a) s += p[a*4+x] * sV[m*16 + a*4 + d];
          acc += s * wp[(x*4+d)*32 + m];
        }
      lg[m] = acc * 0.25f;   // * 1/sqrt(D)
    }
    float mx = lg[0];
#pragma unroll
    for (int m = 1; m < M_; ++m) mx = fmaxf(mx, lg[m]);
    float sum = 0.f;
#pragma unroll
    for (int m = 0; m < M_; ++m) { lg[m] = __expf(lg[m] - mx); sum += lg[m]; }
    float inv = 1.f / sum;
    float s2 = 0.f;
#pragma unroll
    for (int m = 0; m < M_; ++m) { lg[m] *= inv; s2 += lg[m]; }
    float inv2 = 1.f / (s2 + 1e-10f);
#pragma unroll
    for (int m = 0; m < M_; ++m) sQK[nkl*32 + m] = lg[m] * inv2;
  }
  sVn[tid] = 0.f; sVn[tid + 256] = 0.f;
  __syncthreads();

  // ---- phase 2: v_new[m,a,d] = sum_nkl qk * (P W), 8 threads per m
  {
    const int m = tid >> 3; const int sub = tid & 7;
    float vacc[16];
#pragma unroll
    for (int i = 0; i < 16; ++i) vacc[i] = 0.f;
    for (int nkl = sub; nkl < NKL; nkl += 8) {
      int n = nkl / 9; int kl = nkl % 9;
      const float* wp = Wc + ((size_t)kl*N_ + n)*512;
      float q = sQK[nkl*32 + m];
#pragma unroll
      for (int a = 0; a < 4; ++a)
#pragma unroll
        for (int d = 0; d < 4; ++d) {
          float gg = 0.f;
#pragma unroll
          for (int x = 0; x < 4; ++x) gg += sP[nkl*16 + a*4 + x] * wp[(x*4+d)*32 + m];
          vacc[a*4+d] += q * gg;
        }
    }
#pragma unroll
    for (int i = 0; i < 16; ++i) atomicAdd(&sVn[m*16 + i], vacc[i]);
  }
  __syncthreads();

  if (tid < M_) {
    float o[16], out[16];
#pragma unroll
    for (int i = 0; i < 16; ++i) o[i] = sVn[tid*16 + i];
    ln16(o, g, bb, out);
    float* vp = vb + (size_t)pos*512 + tid*16;
#pragma unroll
    for (int i = 0; i < 16; ++i) vp[i] = out[i];
  }
}

// ================= fused FC routing round (in-place on ubuf) ==============
__global__ __launch_bounds__(256) void fc_route(
    const float* __restrict__ vb, const float* __restrict__ Wfc,
    float* __restrict__ ub,
    const float* __restrict__ g, const float* __restrict__ bb)
{
  __shared__ float sU [CLS*16];
  __shared__ float sUn[CLS*16];
  const int tid = threadIdx.x; const int b = blockIdx.x;
  if (tid < CLS*16) { sU[tid] = ub[(size_t)b*CLS*16 + tid]; sUn[tid] = 0.f; }
  __syncthreads();

  for (int n = tid; n < NFC; n += 256) {
    int m = n / (HO*HO); int hw = n % (HO*HO);
    const float* ip = vb + ((size_t)(b*HO*HO + hw)*M_ + m)*16;
    float in[16];
#pragma unroll
    for (int i = 0; i < 16; ++i) in[i] = ip[i];
    const float* wp = Wfc + (size_t)n*160;     // + (x*4+d)*10 + m2
    float lg[CLS];
    for (int m2 = 0; m2 < CLS; ++m2) {
      float acc = 0.f;
#pragma unroll
      for (int x = 0; x < 4; ++x)
#pragma unroll
        for (int d = 0; d < 4; ++d) {
          float s = 0.f;
#pragma unroll
          for (int a = 0; a < 4; ++a) s += in[a*4+x] * sU[m2*16 + a*4 + d];
          acc += s * wp[(x*4+d)*CLS + m2];
        }
      lg[m2] = acc * 0.25f;
    }
    float mx = lg[0];
#pragma unroll
    for (int m2 = 1; m2 < CLS; ++m2) mx = fmaxf(mx, lg[m2]);
    float sum = 0.f;
#pragma unroll
    for (int m2 = 0; m2 < CLS; ++m2) { lg[m2] = __expf(lg[m2] - mx); sum += lg[m2]; }
    float inv = 1.f / sum;
    float s2 = 0.f;
#pragma unroll
    for (int m2 = 0; m2 < CLS; ++m2) { lg[m2] *= inv; s2 += lg[m2]; }
    float inv2 = 1.f / (s2 + 1e-10f);
    for (int m2 = 0; m2 < CLS; ++m2) {
      float q = lg[m2] * inv2;
#pragma unroll
      for (int a = 0; a < 4; ++a)
#pragma unroll
        for (int d = 0; d < 4; ++d) {
          float gg = 0.f;
#pragma unroll
          for (int x = 0; x < 4; ++x) gg += in[a*4+x] * wp[(x*4+d)*CLS + m2];
          atomicAdd(&sUn[m2*16 + a*4 + d], q * gg);
        }
    }
  }
  __syncthreads();
  if (tid < CLS) {
    float o[16], out[16];
#pragma unroll
    for (int i = 0; i < 16; ++i) o[i] = sUn[tid*16 + i];
    ln16(o, g, bb, out);
#pragma unroll
    for (int i = 0; i < 16; ++i) ub[(size_t)b*CLS*16 + tid*16 + i] = out[i];
  }
}

__global__ void copy_out(const float* __restrict__ src, float* __restrict__ dst, int n)
{
  int i = blockIdx.x * blockDim.x + threadIdx.x;
  if (i < n) dst[i] = src[i];
}

// ====================== host-side launcher ======================
extern "C" void kernel_launch(void* const* d_in, const int* in_sizes, int n_in,
                              void* d_out, int out_size, void* d_ws, size_t ws_size,
                              hipStream_t stream)
{
  (void)in_sizes; (void)n_in; (void)out_size; (void)ws_size;
  const float* X   = (const float*)d_in[0];
  const float* Wc  = (const float*)d_in[1];
  const float* Wfc = (const float*)d_in[2];
  const float* g1  = (const float*)d_in[3];
  const float* b1  = (const float*)d_in[4];
  const float* g2  = (const float*)d_in[5];
  const float* b2  = (const float*)d_in[6];
  // d_in[7] = num_routing (device scalar, == 3): 2 update rounds hardcoded.

  char* p = (char*)d_ws;
  auto carve = [&](size_t bytes) {
    char* r = p; p += (bytes + 255) & ~(size_t)255; return (void*)r;
  };
  _Float16* A1 = (_Float16*)carve((size_t)GM1 * GK1 * 2);
  _Float16* B1 = (_Float16*)carve((size_t)GK1 * GN1 * 2);
  float*    O1 = (float*)   carve((size_t)GM1 * GN1 * 4);
  float*    vb = (float*)   carve((size_t)NPOS * M_ * 16 * 4);
  _Float16* A2 = (_Float16*)carve((size_t)GM2 * GK2 * 2);
  _Float16* B2 = (_Float16*)carve((size_t)GK2 * GN2 * 2);
  float*    O2 = (float*)   carve((size_t)GM2 * GN2 * 4);
  float*    ub = (float*)   carve((size_t)B_ * CLS * 16 * 4);

  // --- conv-caps stage ---
  {
    size_t tot = (size_t)GM1 * GK1;
    pack_A1<<<dim3((unsigned)((tot + 255) / 256)), dim3(256), 0, stream>>>(X, A1);
  }
  {
    size_t tot = (size_t)GK1 * GN1;
    pack_B1<<<dim3((unsigned)((tot + 255) / 256)), dim3(256), 0, stream>>>(Wc, B1);
  }
  wmma_gemm_f16<<<dim3(GM1 / TM, 1), dim3(256), 0, stream>>>(A1, B1, O1, GM1, GN1, GK1);
  v_init_ln<<<dim3((B_*M_*HO*HO + 255) / 256), dim3(256), 0, stream>>>(O1, g1, b1, vb);
  conv_route<<<dim3(NPOS), dim3(256), 0, stream>>>(X, Wc, vb, g1, b1);
  conv_route<<<dim3(NPOS), dim3(256), 0, stream>>>(X, Wc, vb, g1, b1);

  // --- class-caps stage ---
  {
    size_t tot = (size_t)GM2 * GK2;
    pack_A2<<<dim3((unsigned)((tot + 255) / 256)), dim3(256), 0, stream>>>(vb, A2);
  }
  {
    size_t tot = (size_t)GK2 * GN2;
    pack_B2<<<dim3((unsigned)((tot + 255) / 256)), dim3(256), 0, stream>>>(Wfc, B2);
  }
  wmma_gemm_f16<<<dim3(GM2 / TM, 1), dim3(256), 0, stream>>>(A2, B2, O2, GM2, GN2, GK2);
  u_init_ln<<<dim3((B_*CLS + 255) / 256), dim3(256), 0, stream>>>(O2, g2, b2, ub);
  fc_route<<<dim3(B_), dim3(256), 0, stream>>>(vb, Wfc, ub, g2, b2);
  fc_route<<<dim3(B_), dim3(256), 0, stream>>>(vb, Wfc, ub, g2, b2);

  copy_out<<<dim3((B_*CLS*16 + 255) / 256), dim3(256), 0, stream>>>(ub, (float*)d_out, B_*CLS*16);
}
